// CrossAttention_20100446945695
// MI455X (gfx1250) — matrix operand order repaired and checked by reading, not compiled, not run on hardware
//
#include <hip/hip_runtime.h>
#include <hip/hip_bf16.h>

// ---------------- CDNA5 WMMA types ----------------
typedef __attribute__((ext_vector_type(16))) __bf16 v16bf;
typedef __attribute__((ext_vector_type(8)))  float  v8f;

#define WMMA_BF16(A, B, C) \
  __builtin_amdgcn_wmma_f32_16x16x32_bf16(false, (A), false, (B), (short)0, (C), false, false)

// ---------------- async global->LDS path (ASYNCcnt) ----------------
#if defined(__has_builtin)
#if __has_builtin(__builtin_amdgcn_global_load_async_to_lds_b128)
#define HAVE_ASYNC_LDS 1
#endif
#endif
#ifndef HAVE_ASYNC_LDS
#define HAVE_ASYNC_LDS 0
#endif

typedef int b128_t __attribute__((vector_size(16)));

static __device__ __forceinline__ void async_copy16(const void* g, void* l) {
#if HAVE_ASYNC_LDS
  __builtin_amdgcn_global_load_async_to_lds_b128(
      (__attribute__((address_space(1))) b128_t*)g,
      (__attribute__((address_space(3))) b128_t*)l, 0, 0);
#else
  *(uint4*)l = *(const uint4*)g;
#endif
}

static __device__ __forceinline__ void async_wait0() {
#if HAVE_ASYNC_LDS
#if defined(__has_builtin) && __has_builtin(__builtin_amdgcn_s_wait_asynccnt)
  __builtin_amdgcn_s_wait_asynccnt(0);
#else
  asm volatile("s_wait_asynccnt 0" ::: "memory");
#endif
#endif
}

// ---------------- problem constants ----------------
#define HEADS    16
#define DIM_HEAD 64
#define NQ       2048
#define NKV      2048
#define DMODEL   1024
#define BATCH    2
#define MROWS    (BATCH * NQ)      // 4096 flattened rows
#define ATT_SCALE 0.125f           // 64^-0.5

// LDS strides (elements) chosen 16B-aligned rows + conflict padding
#define ASTR 40    // 32 K + pad  (80 B/row)
#define BSTR 40
#define KSTR 72    // 64 + pad    (144 B/row)

union FragA { v16bf v; unsigned u[8]; };
union FragB { v16bf v; float4  f4[2]; };

static __device__ inline unsigned pack2(__bf16 a, __bf16 b) {
  union { __bf16 h[2]; unsigned u; } x;
  x.h[0] = a; x.h[1] = b;
  return x.u;
}

// =====================================================================
// Tiled WMMA GEMM:  C[M=4096 x N=1024] = A[4096 x 1024] * W[1024 x 1024]
// SRC_BF16:  A is bf16 (attention output, async-DMA'd) vs f32 (embeddings)
// HEADSPLIT: store bf16 to [b,h,seq,dh] (QKV proj) vs f32 rowmajor + bias
// Block: 256 thr (8 waves), tile 128(M) x 64(N); wave tile 32x32.
// =====================================================================
template <bool SRC_BF16, bool HEADSPLIT>
__global__ __launch_bounds__(256) void gemm_wmma(
    const void* __restrict__ Ap, const float* __restrict__ W,
    void* __restrict__ Outp, const float* __restrict__ bias, float scale) {
  __shared__ __bf16 Alds[128 * ASTR];
  __shared__ __bf16 Blds[64 * BSTR];

  const int t    = threadIdx.x;
  const int lane = t & 31;
  const int wid  = t >> 5;
  const int wm   = wid & 3;          // wave M position (0..3)
  const int wn   = wid >> 2;         // wave N position (0..1)
  const int r16  = lane & 15;
  const int half = lane >> 4;
  const int klo  = half * 8;         // A-frag K interleave base
  const int khB  = half * 8;         // B-frag K interleave base
  const int bx   = blockIdx.x;       // N block (x64)
  const int by   = blockIdx.y;       // M block (x128)

  v8f acc[2][2] = {};

  for (int k0 = 0; k0 < DMODEL; k0 += 32) {
    __syncthreads();
    // ---- stage A tile (128 x 32) -> bf16 LDS ----
    if (SRC_BF16) {
      // already bf16: DMA straight into LDS (ASYNCcnt path)
      const __bf16* A = (const __bf16*)Ap;
#pragma unroll
      for (int p = 0; p < 2; ++p) {
        int idx = p * 256 + t;                   // 0..511
        int row = idx >> 2, seg = idx & 3;       // 8 bf16 per copy
        async_copy16(A + (size_t)(by * 128 + row) * DMODEL + k0 + seg * 8,
                     &Alds[row * ASTR + seg * 8]);
      }
      async_wait0();
    } else {
      const float* A = (const float*)Ap;
#pragma unroll
      for (int p = 0; p < 4; ++p) {
        int idx = p * 256 + t;                   // 0..1023
        int row = idx >> 3, seg = idx & 7;       // 4 f32 per load
        float4 v = *(const float4*)(A + (size_t)(by * 128 + row) * DMODEL + k0 + seg * 4);
        uint2 u;
        u.x = pack2((__bf16)v.x, (__bf16)v.y);
        u.y = pack2((__bf16)v.z, (__bf16)v.w);
        *(uint2*)&Alds[row * ASTR + seg * 4] = u;
      }
    }
    // ---- stage B tile (32K x 64N), stored [n][k] so frags read contiguous K ----
#pragma unroll
    for (int p = 0; p < 2; ++p) {
      int idx = p * 256 + t;                     // 0..511
      int k = idx >> 4;
      int n = (idx & 15) * 4;
      float4 w4 = *(const float4*)(W + (size_t)(k0 + k) * DMODEL + bx * 64 + n);
      Blds[(n + 0) * BSTR + k] = (__bf16)w4.x;
      Blds[(n + 1) * BSTR + k] = (__bf16)w4.y;
      Blds[(n + 2) * BSTR + k] = (__bf16)w4.z;
      Blds[(n + 3) * BSTR + k] = (__bf16)w4.w;
    }
    __syncthreads();

    // ---- fragments + 4 WMMAs ----
    FragA a[2];
#pragma unroll
    for (int i = 0; i < 2; ++i) {
      const __bf16* ar = &Alds[(wm * 32 + i * 16 + r16) * ASTR];
#pragma unroll
      for (int j = 0; j < 4; ++j) {
        a[i].u[j]     = *(const unsigned*)(ar + klo + 2 * j);
        a[i].u[4 + j] = *(const unsigned*)(ar + 16 + klo + 2 * j);
      }
    }
#pragma unroll
    for (int j = 0; j < 2; ++j) {
      FragB bf;
      const __bf16* br = &Blds[(wn * 32 + j * 16 + r16) * BSTR + khB];
      bf.f4[0] = *(const float4*)(br);
      bf.f4[1] = *(const float4*)(br + 16);
#pragma unroll
      for (int i = 0; i < 2; ++i)
        acc[i][j] = WMMA_BF16(a[i].v, bf.v, acc[i][j]);
    }
  }

  // ---- epilogue ----
#pragma unroll
  for (int i = 0; i < 2; ++i)
#pragma unroll
    for (int j = 0; j < 2; ++j)
#pragma unroll
      for (int r = 0; r < 8; ++r) {
        int gm = by * 128 + wm * 32 + i * 16 + r + 8 * half;
        int c  = bx * 64 + wn * 32 + j * 16 + r16;
        float v = acc[i][j][r] * scale;
        if (HEADSPLIT) {
          __bf16* O = (__bf16*)Outp;
          int b = gm >> 11, seq = gm & (NQ - 1);
          int h = c >> 6, dh = c & 63;
          O[(((size_t)(b * HEADS + h)) * NKV + seq) * DIM_HEAD + dh] = (__bf16)v;
        } else {
          float* O = (float*)Outp;
          O[(size_t)gm * DMODEL + c] = v + bias[c];
        }
      }
}

// =====================================================================
// Flash attention: block = 64 query rows of one (b,h), 4 waves x 16 rows.
// Q,K,V bf16 in [b,h,seq,64]; O bf16 row-major [4096 x 1024].
// K tile staged via async global->LDS DMA; V staged manually (transpose).
// =====================================================================
__global__ __launch_bounds__(128) void attn_wmma(
    const __bf16* __restrict__ Q, const __bf16* __restrict__ K,
    const __bf16* __restrict__ V, __bf16* __restrict__ O) {
  __shared__ __bf16 Klds[64 * KSTR];        // [key][dh]
  __shared__ __bf16 Vt[64 * KSTR];          // [dh][key]  (transposed)
  __shared__ __bf16 Plds[4 * 16 * KSTR];    // per-wave P tile [16][64]

  const int t    = threadIdx.x;
  const int lane = t & 31;
  const int w    = t >> 5;
  const int r16  = lane & 15;
  const int half = lane >> 4;
  const int klo  = half * 8;
  const int kh   = half * 8;
  const int b  = blockIdx.z;
  const int h  = blockIdx.y;
  const int q0 = blockIdx.x * 64 + w * 16;

  const size_t bh = (size_t)(b * HEADS + h);
  const __bf16* Qb = Q + bh * NQ * DIM_HEAD;
  const __bf16* Kb = K + bh * NKV * DIM_HEAD;
  const __bf16* Vb = V + bh * NKV * DIM_HEAD;
  __bf16* Pw = &Plds[w * 16 * KSTR];

  // Q A-fragments (loaded once; scale already folded in at projection)
  FragA qf[2];
  {
    const __bf16* qrow = Qb + (size_t)(q0 + r16) * DIM_HEAD;
#pragma unroll
    for (int f = 0; f < 2; ++f)
#pragma unroll
      for (int j = 0; j < 4; ++j) {
        qf[f].u[j]     = *(const unsigned*)(qrow + f * 32 + klo + 2 * j);
        qf[f].u[4 + j] = *(const unsigned*)(qrow + f * 32 + 16 + klo + 2 * j);
      }
  }

  float m8[8], l8[8];
  v8f oacc[4] = {};
#pragma unroll
  for (int r = 0; r < 8; ++r) { m8[r] = -__builtin_inff(); l8[r] = 0.0f; }

  for (int kc = 0; kc < NKV; kc += 64) {
    __syncthreads();
    // ---- stage K tile (async DMA) and transposed V tile (manual) ----
#pragma unroll
    for (int p = 0; p < 4; ++p) {
      int idx = p * 128 + t;                   // 0..511
      int row = idx >> 3, seg = idx & 7;
      async_copy16(Kb + (size_t)(kc + row) * DIM_HEAD + seg * 8,
                   &Klds[row * KSTR + seg * 8]);
      uint4 vv = *(const uint4*)(Vb + (size_t)(kc + row) * DIM_HEAD + seg * 8);
      const __bf16* ve = (const __bf16*)&vv;
#pragma unroll
      for (int e = 0; e < 8; ++e) Vt[(seg * 8 + e) * KSTR + row] = ve[e];
    }
    if (kc + 64 < NKV) {   // global_prefetch_b8 of next chunk
      __builtin_prefetch(Kb + (size_t)(kc + 64 + (t >> 1)) * DIM_HEAD, 0, 1);
      __builtin_prefetch(Vb + (size_t)(kc + 64 + (t >> 1)) * DIM_HEAD, 0, 1);
    }
    async_wait0();
    __syncthreads();

    // ---- S = Q * K^T  (16 x 64 per wave, 8 WMMAs) ----
    v8f sacc[4] = {};
#pragma unroll
    for (int j = 0; j < 4; ++j) {
#pragma unroll
      for (int f = 0; f < 2; ++f) {
        FragB bk;
        const __bf16* kp = &Klds[(j * 16 + r16) * KSTR + f * 32 + kh];
        bk.f4[0] = *(const float4*)kp;
        bk.f4[1] = *(const float4*)(kp + 16);
        sacc[j] = WMMA_BF16(qf[f].v, bk.v, sacc[j]);
      }
    }

    // ---- online softmax (per-row stats; rows live as (r, lane-half)) ----
#pragma unroll
    for (int r = 0; r < 8; ++r) {
      float mx = fmaxf(fmaxf(sacc[0][r], sacc[1][r]), fmaxf(sacc[2][r], sacc[3][r]));
#pragma unroll
      for (int msk = 1; msk < 16; msk <<= 1) mx = fmaxf(mx, __shfl_xor(mx, msk, 32));
      float mnew  = fmaxf(m8[r], mx);
      float alpha = __expf(m8[r] - mnew);
      float ssum  = 0.0f;
#pragma unroll
      for (int j = 0; j < 4; ++j) {
        float p = __expf(sacc[j][r] - mnew);
        ssum += p;
        Pw[(r + 8 * half) * KSTR + j * 16 + r16] = (__bf16)p;
      }
#pragma unroll
      for (int msk = 1; msk < 16; msk <<= 1) ssum += __shfl_xor(ssum, msk, 32);
      l8[r] = l8[r] * alpha + ssum;
      m8[r] = mnew;
#pragma unroll
      for (int j = 0; j < 4; ++j) oacc[j][r] *= alpha;
    }

    // wave-local LDS ordering: P written by all lanes, re-read in A layout
    asm volatile("s_wait_dscnt 0" ::: "memory");
    __builtin_amdgcn_wave_barrier();

    // ---- O += P * V  (8 WMMAs) ----
    FragA pf[2];
    {
      const __bf16* prow = &Pw[r16 * KSTR];
#pragma unroll
      for (int f = 0; f < 2; ++f)
#pragma unroll
        for (int j = 0; j < 4; ++j) {
          pf[f].u[j]     = *(const unsigned*)(prow + f * 32 + klo + 2 * j);
          pf[f].u[4 + j] = *(const unsigned*)(prow + f * 32 + 16 + klo + 2 * j);
        }
    }
#pragma unroll
    for (int j = 0; j < 4; ++j) {
#pragma unroll
      for (int f = 0; f < 2; ++f) {
        FragB bv;
        const __bf16* vp = &Vt[(j * 16 + r16) * KSTR + f * 32 + kh];
        bv.f4[0] = *(const float4*)vp;
        bv.f4[1] = *(const float4*)(vp + 16);
        oacc[j] = WMMA_BF16(pf[f].v, bv.v, oacc[j]);
      }
    }
  }

  // ---- normalize and store O (bf16, row-major [4096 x 1024]) ----
#pragma unroll
  for (int r = 0; r < 8; ++r) {
    float inv = 1.0f / l8[r];
    int qrow = q0 + r + 8 * half;
    __bf16* orow = O + ((size_t)(b * NQ) + qrow) * DMODEL + h * DIM_HEAD;
#pragma unroll
    for (int j = 0; j < 4; ++j)
      orow[j * 16 + r16] = (__bf16)(oacc[j][r] * inv);
  }
}

// =====================================================================
extern "C" void kernel_launch(void* const* d_in, const int* in_sizes, int n_in,
                              void* d_out, int out_size, void* d_ws, size_t ws_size,
                              hipStream_t stream) {
  const float* patch = (const float*)d_in[0];
  const float* pixel = (const float*)d_in[1];
  const float* Wq    = (const float*)d_in[2];
  const float* Wk    = (const float*)d_in[3];
  const float* Wv    = (const float*)d_in[4];
  const float* Wo    = (const float*)d_in[5];
  const float* bo    = (const float*)d_in[6];
  float* out = (float*)d_out;

  const size_t qkvElems = (size_t)BATCH * HEADS * NKV * DIM_HEAD;  // 4M
  __bf16* Qws = (__bf16*)d_ws;
  __bf16* Kws = Qws + qkvElems;
  __bf16* Vws = Kws + qkvElems;
  __bf16* Ows = Vws + qkvElems;   // 4096 x 1024 bf16

  dim3 gemmGrid(DMODEL / 64, MROWS / 128);   // (16, 32)

  gemm_wmma<false, true><<<gemmGrid, 256, 0, stream>>>(patch, Wq, Qws, nullptr, ATT_SCALE);
  gemm_wmma<false, true><<<gemmGrid, 256, 0, stream>>>(pixel, Wk, Kws, nullptr, 1.0f);
  gemm_wmma<false, true><<<gemmGrid, 256, 0, stream>>>(pixel, Wv, Vws, nullptr, 1.0f);

  attn_wmma<<<dim3(NQ / 64, HEADS, BATCH), 128, 0, stream>>>(Qws, Kws, Vws, Ows);

  gemm_wmma<true, false><<<gemmGrid, 256, 0, stream>>>(Ows, Wo, out, bo, 1.0f);
}
